// Sage_Classifier_17154099380260
// MI455X (gfx1250) — matrix operand order repaired and checked by reading, not compiled, not run on hardware
//
#include <hip/hip_runtime.h>

// ---------------------------------------------------------------------------
// GraphSAGE 2-layer forward for MI455X (gfx1250), bf16 WMMA, fp32 accumulate.
// N=16384, F_IN=512, F_HID=256, F_OUT=40.
// Dominant cost: streaming fp32 adj (1 GiB) twice -> ~92us @ 23.3 TB/s.
// BF16 WMMA (v_wmma_f32_16x16x32_bf16) keeps ~420 GFLOP under that roof.
// A-tiles: fp32 global -> cvt -> double-buffered LDS (16-bit A-layout),
// one barrier per 64-wide K step; B pre-transposed to bf16 so each lane's
// fragment is two contiguous global_load_b128 from L2.
// ---------------------------------------------------------------------------

typedef __attribute__((ext_vector_type(16))) __bf16 v16bf;
typedef __attribute__((ext_vector_type(4)))  __bf16 v4bf;
typedef __attribute__((ext_vector_type(8)))  float  v8f;

__device__ __forceinline__ __bf16 f2bf(float f) {
  union { float f; unsigned u; } v; v.f = f;
  unsigned r = v.u + 0x7FFFu + ((v.u >> 16) & 1u);   // round-to-nearest-even
  unsigned short h = (unsigned short)(r >> 16);
  __bf16 o; __builtin_memcpy(&o, &h, 2);
  return o;
}

union FragU { uint4 q[2]; v16bf v; };

// ---------------------------------------------------------------------------
// Elementwise fp32 -> bf16 (row-major copy)
// ---------------------------------------------------------------------------
__global__ __launch_bounds__(256) void cvt_rm(const float* __restrict__ s,
                                              __bf16* __restrict__ d, long n) {
  long i = (long)blockIdx.x * 256 + threadIdx.x;
  if (i < n) d[i] = f2bf(s[i]);
}

// ---------------------------------------------------------------------------
// Transpose + convert: src fp32 [R][C] -> dst bf16 [Cpad][R], zero pad rows
// ---------------------------------------------------------------------------
__global__ __launch_bounds__(256) void transpose_cvt(const float* __restrict__ s,
                                                     __bf16* __restrict__ d,
                                                     int R, int C, int Cpad) {
  long idx = (long)blockIdx.x * 256 + threadIdx.x;
  long total = (long)Cpad * R;
  if (idx >= total) return;
  int c = (int)(idx / R);
  int r = (int)(idx % R);
  d[idx] = (c < C) ? f2bf(s[(size_t)r * C + c]) : f2bf(0.0f);
}

// ---------------------------------------------------------------------------
// GEMM 1: C[M][BN] = A_fp32[M][K] @ B (B given transposed bf16 [BN][K]).
// One block = 32 rows x BN cols (full width) so adj is streamed exactly once.
// Double-buffered LDS A tile, 64-wide K step (2 WMMA sub-steps per stage).
// ---------------------------------------------------------------------------
template <int BN>
__global__ __launch_bounds__(256)
void gemm_adj_bf16(const float* __restrict__ A, const __bf16* __restrict__ BT,
                   __bf16* __restrict__ outRM, int K) {
  constexpr int NT   = BN / 64;        // 16-wide N tiles per wave (4 waves in N)
  constexpr int LSTR = 72;             // bf16 row stride: 144 B, 16B-aligned
  __shared__ __bf16 ldsA[2][32 * LSTR];

  const int tid  = threadIdx.x;
  const int lane = tid & 31;
  const int w    = tid >> 5;
  const int wr   = w >> 2;             // 0..1  (M direction)
  const int wc   = w & 3;              // 0..3  (N direction)
  const int h    = lane >> 4;          // half-wave: selects K sub-range
  const int l16  = lane & 15;
  const int m0   = blockIdx.x * 32;

  v8f zero = {0, 0, 0, 0, 0, 0, 0, 0};
  v8f acc[NT];
#pragma unroll
  for (int t = 0; t < NT; ++t) acc[t] = zero;

  // loader mapping: 256 thr -> 32 rows x 8 col-groups; each thread covers
  // fp32 cols [acg, acg+4) and [acg+32, acg+36) of the 64-wide tile
  const int arow = tid >> 3;
  const int acg  = (tid & 7) * 4;
  const float* aptr = A + (size_t)(m0 + arow) * K + acg;

  float4 f0 = *(const float4*)(aptr);
  float4 f1 = *(const float4*)(aptr + 32);

  int buf = 0;
  for (int kk = 0; kk < K; kk += 64) {
    // convert staged registers -> bf16 LDS tile (A-matrix lane layout source)
    v4bf p0, p1;
    p0[0] = f2bf(f0.x); p0[1] = f2bf(f0.y); p0[2] = f2bf(f0.z); p0[3] = f2bf(f0.w);
    p1[0] = f2bf(f1.x); p1[1] = f2bf(f1.y); p1[2] = f2bf(f1.z); p1[3] = f2bf(f1.w);
    __bf16* lrow = &ldsA[buf][arow * LSTR];
    *(v4bf*)(lrow + acg)      = p0;
    *(v4bf*)(lrow + 32 + acg) = p1;
    __syncthreads();

    // issue next tile's global loads early; they overlap the WMMA block below
    if (kk + 64 < K) {
      f0 = *(const float4*)(aptr + kk + 64);
      f1 = *(const float4*)(aptr + kk + 96);
      __builtin_prefetch(aptr + kk + 128, 0, 1);   // global_prefetch_b8
    }

    const __bf16* ap = &ldsA[buf][(wr * 16 + l16) * LSTR];
#pragma unroll
    for (int ks = 0; ks < 2; ++ks) {
      // A fragment: lane(m=l16,h): 16 contiguous K vals at K[h*16..+15]
      FragU ua;
      ua.q[0] = *(const uint4*)(ap + ks * 32 + h * 16);
      ua.q[1] = *(const uint4*)(ap + ks * 32 + h * 16 + 8);
      v16bf afrag = ua.v;
#pragma unroll
      for (int t = 0; t < NT; ++t) {
        // B fragment: lane(n=l16,h): 16 contiguous K vals at BT[n][kk+ks*32+h*16]
        const __bf16* bp =
            BT + (size_t)(wc * (BN / 4) + t * 16 + l16) * K + kk + ks * 32 + h * 16;
        FragU ub;
        ub.q[0] = *(const uint4*)bp;
        ub.q[1] = *(const uint4*)(bp + 8);
        acc[t] = __builtin_amdgcn_wmma_f32_16x16x32_bf16(
            false, afrag, false, ub.v, (short)0, acc[t], false, false);
      }
    }
    buf ^= 1;
  }

  // C layout: VGPR r holds (m = r + 8*h, n = l16)
#pragma unroll
  for (int t = 0; t < NT; ++t) {
    int ncol = wc * (BN / 4) + t * 16 + l16;
    size_t base = (size_t)(m0 + wr * 16 + 8 * h) * BN + ncol;
#pragma unroll
    for (int r = 0; r < 8; ++r)
      outRM[base + (size_t)r * BN] = f2bf(acc[t][r]);
  }
}

// ---------------------------------------------------------------------------
// GEMM 2: C = act([A0 | A1] @ B + bias). A0: bf16 [M][K0LEN], A1: bf16 [M][K-K0LEN],
// B transposed bf16 [BN][K]. Optional relu, row-major bf16 out, transposed bf16
// out, and fp32 out clipped to NREAL columns. K is small; no LDS needed.
// ---------------------------------------------------------------------------
template <int BN, int K0LEN, bool RELU, bool OUT_RM, bool OUT_TR, bool OUT_F32>
__global__ __launch_bounds__(256)
void gemm_concat_bf16(const __bf16* __restrict__ A0, const __bf16* __restrict__ A1,
                      const __bf16* __restrict__ BT, const float* __restrict__ bias,
                      int K, int M, int NREAL,
                      __bf16* __restrict__ outRM, __bf16* __restrict__ outTR,
                      float* __restrict__ outF32) {
  constexpr int NT = BN / 64;
  const int tid  = threadIdx.x;
  const int lane = tid & 31;
  const int w    = tid >> 5;
  const int wr   = w >> 2;
  const int wc   = w & 3;
  const int h    = lane >> 4;
  const int l16  = lane & 15;
  const int m0   = blockIdx.x * 32;
  const int mrow = m0 + wr * 16 + l16;

  v8f zero = {0, 0, 0, 0, 0, 0, 0, 0};
  v8f acc[NT];
#pragma unroll
  for (int t = 0; t < NT; ++t) acc[t] = zero;

  for (int kk = 0; kk < K; kk += 32) {
    const __bf16* asrc;
    int kloc, ksz;
    if (kk < K0LEN) { asrc = A0; kloc = kk;         ksz = K0LEN;     }
    else            { asrc = A1; kloc = kk - K0LEN; ksz = K - K0LEN; }
    const __bf16* ap = asrc + (size_t)mrow * ksz + kloc;
    FragU ua;
    ua.q[0] = *(const uint4*)(ap + h * 16);
    ua.q[1] = *(const uint4*)(ap + h * 16 + 8);
    v16bf afrag = ua.v;

#pragma unroll
    for (int t = 0; t < NT; ++t) {
      const __bf16* bp = BT + (size_t)(wc * (BN / 4) + t * 16 + l16) * K + kk + h * 16;
      FragU ub;
      ub.q[0] = *(const uint4*)bp;
      ub.q[1] = *(const uint4*)(bp + 8);
      acc[t] = __builtin_amdgcn_wmma_f32_16x16x32_bf16(
          false, afrag, false, ub.v, (short)0, acc[t], false, false);
    }
  }

#pragma unroll
  for (int t = 0; t < NT; ++t) {
    int ncol = wc * (BN / 4) + t * 16 + l16;
    float bv = (ncol < NREAL) ? bias[ncol] : 0.0f;
#pragma unroll
    for (int r = 0; r < 8; ++r) {
      int mr = m0 + wr * 16 + 8 * h + r;
      float v = acc[t][r] + bv;
      if (RELU) v = fmaxf(v, 0.0f);
      if (OUT_RM) outRM[(size_t)mr * BN + ncol] = f2bf(v);
      if (OUT_TR) outTR[(size_t)ncol * M + mr] = f2bf(v);
      if (OUT_F32) { if (ncol < NREAL) outF32[(size_t)mr * NREAL + ncol] = v; }
    }
  }
}

// ---------------------------------------------------------------------------
extern "C" void kernel_launch(void* const* d_in, const int* in_sizes, int n_in,
                              void* d_out, int out_size, void* d_ws, size_t ws_size,
                              hipStream_t stream) {
  const float* adj = (const float*)d_in[0];
  const float* x   = (const float*)d_in[1];
  const float* W1  = (const float*)d_in[2];
  const float* b1  = (const float*)d_in[3];
  const float* W2  = (const float*)d_in[4];
  const float* b2  = (const float*)d_in[5];
  float* out = (float*)d_out;

  constexpr int N = 16384, FIN = 512, FHID = 256, FOUT = 40, FOUTP = 64;

  char* wsp = (char*)d_ws;
  auto carve = [&](size_t bytes) {
    void* p = (void*)wsp;
    wsp += (bytes + 255) & ~(size_t)255;
    return p;
  };
  __bf16* xT  = (__bf16*)carve((size_t)FIN * N * 2);            // x^T   bf16 [512][16384]
  __bf16* xrm = (__bf16*)carve((size_t)N * FIN * 2);            // x     bf16 [16384][512]
  __bf16* c1  = (__bf16*)carve((size_t)N * FIN * 2);            // adj@x bf16 [16384][512]
  __bf16* w1t = (__bf16*)carve((size_t)FHID * (2 * FIN) * 2);   // W1^T  bf16 [256][1024]
  __bf16* hrm = (__bf16*)carve((size_t)N * FHID * 2);           // h     bf16 [16384][256]
  __bf16* hT  = (__bf16*)carve((size_t)FHID * N * 2);           // h^T   bf16 [256][16384]
  __bf16* c2  = (__bf16*)carve((size_t)N * FHID * 2);           // adj@h bf16 [16384][256]
  __bf16* w2t = (__bf16*)carve((size_t)FOUTP * (2 * FHID) * 2); // W2^T padded [64][512]

  dim3 blk(256);

  // operand precision conversion (tiny vs. the 2 GiB adj stream)
  cvt_rm<<<((long)N * FIN + 255) / 256, blk, 0, stream>>>(x, xrm, (long)N * FIN);
  transpose_cvt<<<((long)FIN * N + 255) / 256, blk, 0, stream>>>(x, xT, N, FIN, FIN);
  transpose_cvt<<<((long)FHID * 2 * FIN + 255) / 256, blk, 0, stream>>>(W1, w1t, 2 * FIN, FHID, FHID);
  transpose_cvt<<<((long)FOUTP * 2 * FHID + 255) / 256, blk, 0, stream>>>(W2, w2t, 2 * FHID, FOUT, FOUTP);

  // layer 1: C1 = adj @ x ; h = relu([C1|x] @ W1 + b1)
  gemm_adj_bf16<512><<<N / 32, blk, 0, stream>>>(adj, xT, c1, N);
  gemm_concat_bf16<256, 512, true, true, true, false><<<N / 32, blk, 0, stream>>>(
      c1, xrm, w1t, b1, 2 * FIN, N, FHID, hrm, hT, nullptr);

  // layer 2: C2 = adj @ h ; out = [C2|h] @ W2 + b2
  gemm_adj_bf16<256><<<N / 32, blk, 0, stream>>>(adj, hT, c2, N);
  gemm_concat_bf16<64, 256, false, false, false, true><<<N / 32, blk, 0, stream>>>(
      c2, hrm, w2t, b2, 2 * FHID, N, FOUT, nullptr, nullptr, out);
}
